// Qwen3_5TextModel_75917841924556
// MI455X (gfx1250) — matrix operand order repaired and checked by reading, not compiled, not run on hardware
//
#include <hip/hip_runtime.h>
#include <hip/hip_bf16.h>
#include <math.h>

// ---------------------------------------------------------------------------
// Problem constants: B=1, S=2048, H=2048, NH=16, NKV=4, HD=128, RD=64
// ---------------------------------------------------------------------------
static constexpr int S_  = 2048;
static constexpr int H_  = 2048;
static constexpr int NH_ = 16;
static constexpr int NKV_= 4;
static constexpr int HD_ = 128;
static constexpr int RD_ = 64;
static constexpr float EPS_ = 1e-6f;

typedef __attribute__((ext_vector_type(16))) _Float16 v16h;
typedef __attribute__((ext_vector_type(8)))  _Float16 v8h;
typedef __attribute__((ext_vector_type(8)))  float    v8f;
typedef __attribute__((ext_vector_type(4)))  unsigned int u32x4;
typedef __attribute__((ext_vector_type(8)))  int      i32x8;
typedef __attribute__((ext_vector_type(4)))  int      i32x4;

// ---------------------------------------------------------------------------
// TDM availability detection (ROCm 7.2 = 5-arg builtin; therock/clang-23 ships
// the TDM header and uses the 6-arg builtin).  Fallback: manual LDS staging.
// ---------------------------------------------------------------------------
#if defined(__has_builtin)
#  if __has_builtin(__builtin_amdgcn_tensor_load_to_lds) && \
      __has_builtin(__builtin_amdgcn_s_wait_tensorcnt)
#    define USE_TDM 1
#  endif
#endif
#ifndef USE_TDM
#  define USE_TDM 0
#endif
#if __has_include(<hip/amd_detail/amd_gfx1250_TDM.h>)
#  define TDM_6ARG 1
#else
#  define TDM_6ARG 0
#endif

// ---------------------------------------------------------------------------
// The TDM DMA writes LDS behind the compiler's back.  Escape the LDS address
// into asm (so the shared array is "asm-reachable") and use memory clobbers
// at the sync points so the B-panel ds_loads are really emitted each K-step
// instead of being folded to undef (observed in round-2 asm).
// ---------------------------------------------------------------------------
__device__ __forceinline__ void lds_escape(void* p) {
    asm volatile("" : "+v"(p) : : "memory");
}
__device__ __forceinline__ void mem_clobber() {
    asm volatile("" : : : "memory");
}

// ---------------------------------------------------------------------------
// Stage a 32 x 64 f16 B-panel (rows k0..k0+31, cols n0..n0+63 of a row-major
// [K x N] matrix, row stride ldb) into LDS at offset 0.
// TDM path: wave 0 issues TENSOR_LOAD_TO_LDS with a 2-D D# descriptor and
// waits on TENSORcnt; others just hit the barrier outside.
// ---------------------------------------------------------------------------
__device__ __forceinline__ void stage_b_panel(const _Float16* __restrict__ Bm,
                                              int ldb, int N, int K,
                                              int k0, int n0blk,
                                              _Float16* bs, int tid) {
#if USE_TDM
    if (tid < 32) {
        const unsigned long long ga =
            (unsigned long long)(const void*)(Bm + (size_t)k0 * ldb + n0blk);
        u32x4 g0;
        g0[0] = 1u;                                    // count=1, user D#
        g0[1] = 0u;                                    // lds_addr = 0 (bs base)
        g0[2] = (unsigned int)(ga & 0xFFFFFFFFu);      // global_addr[31:0]
        g0[3] = (unsigned int)((ga >> 32) & 0x1FFFFFFu) | 0x80000000u; // +type=2
        i32x8 g1;
        g1[0] = 0x00010000;                            // data_size=1 (2 bytes)
        g1[1] = (int)((N & 0xFFFF) << 16);             // tensor_dim0 lo
        g1[2] = (int)((N >> 16) & 0xFFFF) | ((K & 0xFFFF) << 16); // dim0 hi|dim1 lo
        g1[3] = (int)((K >> 16) & 0xFFFF) | (64 << 16);// dim1 hi | tile_dim0=64
        g1[4] = 32;                                    // tile_dim1=32, tile_dim2=0
        g1[5] = N;                                     // tensor_dim0_stride lo
        g1[6] = 0;                                     // stride hi | dim1_stride lo
        g1[7] = 0;
        i32x4 gz; gz[0] = 0; gz[1] = 0; gz[2] = 0; gz[3] = 0;
#if TDM_6ARG
        i32x8 gz8;
        for (int i = 0; i < 8; ++i) gz8[i] = 0;
        __builtin_amdgcn_tensor_load_to_lds(g0, g1, gz, gz, gz8, 0);
#else
        __builtin_amdgcn_tensor_load_to_lds(g0, g1, gz, gz, 0);
#endif
        __builtin_amdgcn_s_wait_tensorcnt(0);
        mem_clobber();   // LDS contents changed by DMA
    }
    (void)bs;
#else
    const int row = tid >> 3;             // 0..31
    const int col = (tid & 7) * 8;        // 0..56
    *(v8h*)(bs + row * 64 + col) =
        *(const v8h*)(Bm + (size_t)(k0 + row) * ldb + n0blk + col);
#endif
}

// ---------------------------------------------------------------------------
// Core block GEMM: 256 threads (8 waves).  Block computes a 128 x 64 C-tile;
// wave w handles rows m0 = blockRow + 16*w against 4 accumulators (N-tiles).
// Fragment layouts per CDNA5 ISA 7.12.2:
//   A 16x32 f16: lanes 0-15 M=lane / Kbase 0; lanes 16-31 M=lane-16 / Kbase 8;
//                halves 0..7 -> K=base..base+7, 8..15 -> K=base+16..base+23.
//   B 32x16 f16: lane L holds column N=L&15, Kbase 8*(L>=16); halves -> K as for A.
//   C/D 16x16 f32: VGPR r -> M = r + 8*(lane>=16), N = lane&15.
// ---------------------------------------------------------------------------
__device__ __forceinline__ void gemm_core(const _Float16* __restrict__ A, int lda,
                                          const _Float16* __restrict__ Bm, int ldb,
                                          int N, int K, int m0, int n0blk,
                                          _Float16* bs, v8f acc[4]) {
    const int tid  = threadIdx.x;
    const int lane = tid & 31;
    const _Float16* arow =
        A + (size_t)(m0 + (lane & 15)) * lda + ((lane >> 4) << 3);
    for (int k0 = 0; k0 < K; k0 += 32) {
        stage_b_panel(Bm, ldb, N, K, k0, n0blk, bs, tid);
        if (k0 + 32 < K)   // speculative prefetch of next A chunk
            __builtin_prefetch(arow + k0 + 32, 0, 1);
        __syncthreads();
        mem_clobber();     // B panel in LDS was (re)written before this point
        const v8h alo = *(const v8h*)(arow + k0);
        const v8h ahi = *(const v8h*)(arow + k0 + 16);
        const v16h a = __builtin_shufflevector(alo, ahi,
            0, 1, 2, 3, 4, 5, 6, 7, 8, 9, 10, 11, 12, 13, 14, 15);
#pragma unroll
        for (int t = 0; t < 4; ++t) {
            v16h b;
#pragma unroll
            for (int s = 0; s < 16; ++s)
                b[s] = bs[(((s & 8) << 1) + ((lane >> 4) << 3) + (s & 7)) * 64 + t * 16 + (lane & 15)];
            acc[t] = __builtin_amdgcn_wmma_f32_16x16x32_f16(
                false, a, false, b, (short)0, acc[t], false, false);
        }
        __syncthreads();
    }
}

// ---------------------------------------------------------------------------
// Kernel: generic f16 GEMM, f32 store.  grid = (N/64, M/128), block = 256
// ---------------------------------------------------------------------------
__global__ void k_gemm16(const _Float16* __restrict__ A, int lda,
                         const _Float16* __restrict__ Bm, int ldb, int N,
                         float* __restrict__ C, int ldc, int K) {
    __shared__ __align__(32) _Float16 bs[32 * 64];
    lds_escape(bs);
    const int tid  = threadIdx.x;
    const int lane = tid & 31;
    const int n0blk = blockIdx.x * 64;
    const int m0    = blockIdx.y * 128 + (tid >> 5) * 16;
    v8f acc[4];
#pragma unroll
    for (int t = 0; t < 4; ++t)
#pragma unroll
        for (int r = 0; r < 8; ++r) acc[t][r] = 0.0f;
    gemm_core(A, lda, Bm, ldb, N, K, m0, n0blk, bs, acc);
    const int nl = lane & 15;
    const int mb = m0 + ((lane >> 4) << 3);
#pragma unroll
    for (int t = 0; t < 4; ++t)
#pragma unroll
        for (int r = 0; r < 8; ++r)
            C[(size_t)(mb + r) * ldc + n0blk + t * 16 + nl] = acc[t][r];
}

// ---------------------------------------------------------------------------
// Kernel: f32 -> f16 streaming convert
// ---------------------------------------------------------------------------
__global__ void k_cvt(const float* __restrict__ src, _Float16* __restrict__ dst,
                      int n) {
    for (int i = blockIdx.x * 256 + threadIdx.x; i < n; i += gridDim.x * 256)
        dst[i] = (_Float16)src[i];
}

// ---------------------------------------------------------------------------
// Q RMSNorm + RoPE: q_raw [s][h][2*HD] (query|gate) -> qfh [h][s][d] (f16)
// grid = (S, NH), block = HD
// ---------------------------------------------------------------------------
__global__ void k_q_normrope(const float* __restrict__ qraw,
                             const float* __restrict__ cosb,
                             const float* __restrict__ sinb,
                             const float* __restrict__ qw,
                             _Float16* __restrict__ qf) {
    const int s = blockIdx.x, h = blockIdx.y, d = threadIdx.x;
    __shared__ float ssq[HD_];
    __shared__ float sy[HD_];
    const float v = qraw[(size_t)s * (NH_ * 2 * HD_) + h * (2 * HD_) + d];
    ssq[d] = v * v;
    __syncthreads();
    for (int off = HD_ / 2; off > 0; off >>= 1) {
        if (d < off) ssq[d] += ssq[d + off];
        __syncthreads();
    }
    const float inv = rsqrtf(ssq[0] * (1.0f / HD_) + EPS_);
    const float y = v * inv * qw[d];
    sy[d] = y;
    __syncthreads();
    float o = y;
    if (d < RD_) {
        const float c  = cosb[(size_t)s * RD_ + d];
        const float sn = sinb[(size_t)s * RD_ + d];
        const float rot = (d < RD_ / 2) ? -sy[d + RD_ / 2] : sy[d - RD_ / 2];
        o = y * c + rot * sn;
    }
    qf[((size_t)h * S_ + s) * HD_ + d] = (_Float16)o;
}

// ---------------------------------------------------------------------------
// K RMSNorm + RoPE, stored TRANSPOSED: kfh [kv][d][s] (f16).
// grid = (S, NKV), block = HD
// ---------------------------------------------------------------------------
__global__ void k_k_normrope(const float* __restrict__ kraw,
                             const float* __restrict__ cosb,
                             const float* __restrict__ sinb,
                             const float* __restrict__ kw,
                             _Float16* __restrict__ kf) {
    const int s = blockIdx.x, kv = blockIdx.y, d = threadIdx.x;
    __shared__ float ssq[HD_];
    __shared__ float sy[HD_];
    const float v = kraw[(size_t)s * (NKV_ * HD_) + kv * HD_ + d];
    ssq[d] = v * v;
    __syncthreads();
    for (int off = HD_ / 2; off > 0; off >>= 1) {
        if (d < off) ssq[d] += ssq[d + off];
        __syncthreads();
    }
    const float inv = rsqrtf(ssq[0] * (1.0f / HD_) + EPS_);
    const float y = v * inv * kw[d];
    sy[d] = y;
    __syncthreads();
    float o = y;
    if (d < RD_) {
        const float c  = cosb[(size_t)s * RD_ + d];
        const float sn = sinb[(size_t)s * RD_ + d];
        const float rot = (d < RD_ / 2) ? -sy[d + RD_ / 2] : sy[d - RD_ / 2];
        o = y * c + rot * sn;
    }
    kf[((size_t)kv * HD_ + d) * S_ + s] = (_Float16)o;
}

// ---------------------------------------------------------------------------
// V relayout + convert: vraw [s][kv*HD+d] f32 -> vfh [kv][s][d] f16
// ---------------------------------------------------------------------------
__global__ void k_v_copy(const float* __restrict__ vraw,
                         _Float16* __restrict__ vf) {
    const size_t i = (size_t)blockIdx.x * blockDim.x + threadIdx.x;
    if (i >= (size_t)S_ * NKV_ * HD_) return;
    const int s = (int)(i / (NKV_ * HD_));
    const int j = (int)(i % (NKV_ * HD_));
    const int kv = j / HD_, d = j % HD_;
    vf[((size_t)kv * S_ + s) * HD_ + d] = (_Float16)vraw[i];
}

// ---------------------------------------------------------------------------
// Attention scores: attn[h][q][k] = scaling * Q_h @ K_h^T.
// grid = (S/64, S/128, NH), block = 256.  K-dim = HD (4 WMMA steps).
// ---------------------------------------------------------------------------
__global__ void k_scores(const _Float16* __restrict__ qf,
                         const _Float16* __restrict__ kf,
                         float* __restrict__ attn, float scaling) {
    __shared__ __align__(32) _Float16 bs[32 * 64];
    lds_escape(bs);
    const int tid  = threadIdx.x;
    const int lane = tid & 31;
    const int h    = blockIdx.z;
    const int kv   = h / (NH_ / NKV_);
    const int n0blk = blockIdx.x * 64;                    // key index
    const int m0    = blockIdx.y * 128 + (tid >> 5) * 16; // query index
    const _Float16* A  = qf + (size_t)h * S_ * HD_;       // [S][HD]
    const _Float16* Bm = kf + (size_t)kv * HD_ * S_;      // [HD][S]
    v8f acc[4];
#pragma unroll
    for (int t = 0; t < 4; ++t)
#pragma unroll
        for (int r = 0; r < 8; ++r) acc[t][r] = 0.0f;
    gemm_core(A, HD_, Bm, S_, S_, HD_, m0, n0blk, bs, acc);
    float* C = attn + (size_t)h * S_ * S_;
    const int nl = lane & 15;
    const int mb = m0 + ((lane >> 4) << 3);
#pragma unroll
    for (int t = 0; t < 4; ++t)
#pragma unroll
        for (int r = 0; r < 8; ++r)
            C[(size_t)(mb + r) * S_ + n0blk + t * 16 + nl] = scaling * acc[t][r];
}

// ---------------------------------------------------------------------------
// Masked softmax (reference semantics: masked entries take rowmin-20), writes
// f32 probs in place (this is the required attn output) + f16 copy for P@V.
// grid = (S, NH), block = 256
// ---------------------------------------------------------------------------
__global__ void k_softmax(float* __restrict__ attn,
                          const int* __restrict__ mask,
                          _Float16* __restrict__ probsh) {
    const int q = blockIdx.x, h = blockIdx.y, tid = threadIdx.x;
    const int NT = 256;
    __shared__ float sb[256];
    float*     P   = attn   + ((size_t)h * S_ + q) * S_;
    _Float16*  P16 = probsh + ((size_t)h * S_ + q) * S_;
    const int* Mr  = mask + (size_t)q * S_;

    float lmin = 3.4e38f;
    for (int k = tid; k < S_; k += NT) lmin = fminf(lmin, P[k]);
    sb[tid] = lmin; __syncthreads();
    for (int off = NT / 2; off > 0; off >>= 1) {
        if (tid < off) sb[tid] = fminf(sb[tid], sb[tid + off]);
        __syncthreads();
    }
    const float vv = sb[0] - 20.0f;
    __syncthreads();

    float lmax = -3.4e38f;
    for (int k = tid; k < S_; k += NT) {
        const float val = (Mr[k] == 0) ? P[k] : vv;
        lmax = fmaxf(lmax, val);
    }
    sb[tid] = lmax; __syncthreads();
    for (int off = NT / 2; off > 0; off >>= 1) {
        if (tid < off) sb[tid] = fmaxf(sb[tid], sb[tid + off]);
        __syncthreads();
    }
    const float rmax = sb[0];
    __syncthreads();

    float lsum = 0.0f;
    for (int k = tid; k < S_; k += NT) {
        const float val = (Mr[k] == 0) ? P[k] : vv;
        lsum += __expf(val - rmax);
    }
    sb[tid] = lsum; __syncthreads();
    for (int off = NT / 2; off > 0; off >>= 1) {
        if (tid < off) sb[tid] += sb[tid + off];
        __syncthreads();
    }
    const float rinv = 1.0f / sb[0];
    __syncthreads();

    for (int k = tid; k < S_; k += NT) {
        const float val = (Mr[k] == 0) ? P[k] : vv;
        const float p = __expf(val - rmax) * rinv;
        P[k]   = p;
        P16[k] = (_Float16)p;
    }
}

// ---------------------------------------------------------------------------
// P@V with sigmoid gating: og[s][h*HD+d] (f16) = (P_h @ V_kv) * sig(gate).
// grid = (HD/64, S/128, NH), block = 256.  K-dim = S.
// ---------------------------------------------------------------------------
__global__ void k_av(const _Float16* __restrict__ probsh,
                     const _Float16* __restrict__ vf,
                     const float* __restrict__ qraw,   // gates live here
                     _Float16* __restrict__ og) {
    __shared__ __align__(32) _Float16 bs[32 * 64];
    lds_escape(bs);
    const int tid  = threadIdx.x;
    const int lane = tid & 31;
    const int h    = blockIdx.z;
    const int kv   = h / (NH_ / NKV_);
    const int n0blk = blockIdx.x * 64;                    // head dim
    const int m0    = blockIdx.y * 128 + (tid >> 5) * 16; // query index
    const _Float16* A  = probsh + (size_t)h * S_ * S_;    // [S][S]
    const _Float16* Bm = vf     + (size_t)kv * S_ * HD_;  // [S][HD]
    v8f acc[4];
#pragma unroll
    for (int t = 0; t < 4; ++t)
#pragma unroll
        for (int r = 0; r < 8; ++r) acc[t][r] = 0.0f;
    gemm_core(A, S_, Bm, HD_, HD_, S_, m0, n0blk, bs, acc);
    const int nl = lane & 15;
    const int mb = m0 + ((lane >> 4) << 3);
#pragma unroll
    for (int t = 0; t < 4; ++t)
#pragma unroll
        for (int r = 0; r < 8; ++r) {
            const int m = mb + r;
            const int n = n0blk + t * 16 + nl;
            const float g =
                qraw[(size_t)m * (NH_ * 2 * HD_) + h * (2 * HD_) + HD_ + n];
            const float sig = 1.0f / (1.0f + __expf(-g));
            og[(size_t)m * (NH_ * HD_) + h * HD_ + n] =
                (_Float16)(acc[t][r] * sig);
        }
}

// ---------------------------------------------------------------------------
// Host-side launch
// ---------------------------------------------------------------------------
extern "C" void kernel_launch(void* const* d_in, const int* in_sizes, int n_in,
                              void* d_out, int out_size, void* d_ws, size_t ws_size,
                              hipStream_t stream) {
    (void)in_sizes; (void)n_in; (void)out_size; (void)ws_size;
    const float* x    = (const float*)d_in[0];
    const float* cosb = (const float*)d_in[1];
    const float* sinb = (const float*)d_in[2];
    const int*   mask = (const int*)  d_in[3];
    const float* wq   = (const float*)d_in[4];
    const float* wk   = (const float*)d_in[5];
    const float* wv   = (const float*)d_in[6];
    const float* wo   = (const float*)d_in[7];
    const float* qnw  = (const float*)d_in[8];
    const float* knw  = (const float*)d_in[9];

    float* out  = (float*)d_out;                 // (S,H)
    float* attn = out + (size_t)S_ * H_;         // (NH,S,S)

    // workspace layout
    char* p = (char*)d_ws;
    auto alloc_f = [&](size_t n) { float* r = (float*)p;    p += n * 4; return r; };
    auto alloc_h = [&](size_t n) { _Float16* r = (_Float16*)p; p += n * 2; return r; };
    float*    qraw  = alloc_f((size_t)S_ * NH_ * 2 * HD_);  // 32 MB
    float*    kraw  = alloc_f((size_t)S_ * NKV_ * HD_);     //  4 MB
    float*    vraw  = alloc_f((size_t)S_ * NKV_ * HD_);     //  4 MB
    _Float16* xh    = alloc_h((size_t)S_ * H_);             //  8 MB
    _Float16* wqh   = alloc_h((size_t)H_ * NH_ * 2 * HD_);  // 16 MB
    _Float16* wkh   = alloc_h((size_t)H_ * NKV_ * HD_);     //  2 MB
    _Float16* wvh   = alloc_h((size_t)H_ * NKV_ * HD_);     //  2 MB
    _Float16* woh   = alloc_h((size_t)NH_ * HD_ * H_);      //  8 MB
    _Float16* qfh   = alloc_h((size_t)NH_ * S_ * HD_);      //  8 MB
    _Float16* kfh   = alloc_h((size_t)NKV_ * HD_ * S_);     //  2 MB
    _Float16* vfh   = alloc_h((size_t)NKV_ * S_ * HD_);     //  2 MB
    _Float16* ogh   = alloc_h((size_t)S_ * NH_ * HD_);      //  8 MB
    _Float16* prb   = alloc_h((size_t)NH_ * S_ * S_);       // 128 MB

    const float scaling = 0.08838834764831845f;  // HD^-0.5

    // 0) one-shot f32 -> f16 operand conversion (streaming, bandwidth bound)
    k_cvt<<<2048, 256, 0, stream>>>(x,  xh,  S_ * H_);
    k_cvt<<<2048, 256, 0, stream>>>(wq, wqh, H_ * NH_ * 2 * HD_);
    k_cvt<<<2048, 256, 0, stream>>>(wk, wkh, H_ * NKV_ * HD_);
    k_cvt<<<2048, 256, 0, stream>>>(wv, wvh, H_ * NKV_ * HD_);
    k_cvt<<<2048, 256, 0, stream>>>(wo, woh, NH_ * HD_ * H_);

    // 1) projections (f16 operands, f32 accum/out)
    k_gemm16<<<dim3((NH_ * 2 * HD_) / 64, S_ / 128), 256, 0, stream>>>(
        xh, H_, wqh, NH_ * 2 * HD_, NH_ * 2 * HD_, qraw, NH_ * 2 * HD_, H_);
    k_gemm16<<<dim3((NKV_ * HD_) / 64, S_ / 128), 256, 0, stream>>>(
        xh, H_, wkh, NKV_ * HD_, NKV_ * HD_, kraw, NKV_ * HD_, H_);
    k_gemm16<<<dim3((NKV_ * HD_) / 64, S_ / 128), 256, 0, stream>>>(
        xh, H_, wvh, NKV_ * HD_, NKV_ * HD_, vraw, NKV_ * HD_, H_);

    // 2) norms + rope + relayout (f16 outputs)
    k_q_normrope<<<dim3(S_, NH_), HD_, 0, stream>>>(qraw, cosb, sinb, qnw, qfh);
    k_k_normrope<<<dim3(S_, NKV_), HD_, 0, stream>>>(kraw, cosb, sinb, knw, kfh);
    k_v_copy<<<dim3((S_ * NKV_ * HD_ + 255) / 256), 256, 0, stream>>>(vraw, vfh);

    // 3) scores -> attn region of d_out, masked softmax in place (+f16 copy)
    k_scores<<<dim3(S_ / 64, S_ / 128, NH_), 256, 0, stream>>>(qfh, kfh, attn, scaling);
    k_softmax<<<dim3(S_, NH_), 256, 0, stream>>>(attn, mask, prb);

    // 4) P@V with sigmoid gating (f16 out)
    k_av<<<dim3(HD_ / 64, S_ / 128, NH_), 256, 0, stream>>>(prb, vfh, qraw, ogh);

    // 5) output projection: out = og @ wo
    k_gemm16<<<dim3(H_ / 64, S_ / 128), 256, 0, stream>>>(
        ogh, NH_ * HD_, woh, H_, H_, out, H_, NH_ * HD_);
}
